// BiDAFNet_26525718020551
// MI455X (gfx1250) — matrix operand order repaired and checked by reading, not compiled, not run on hardware
//
#include <hip/hip_runtime.h>
#include <hip/hip_bf16.h>

// ---------------------------------------------------------------------------
// BiDAF attention flow for MI455X (gfx1250, wave32, WMMA bf16).
//   B=8, XL=YL=2048, H=768, OUT=300
// Dominant cost: two 2048x2048x768 batched GEMMs -> v_wmma_f32_16x16x32_bf16.
// Register-blocked: each wave computes a 32x32 macro-tile (4 WMMA accumulators),
// so each K-step issues 4 fragment loads for 4 WMMAs (2 loads/WMMA vs 4).
// ---------------------------------------------------------------------------

#define B_    8
#define XL_   2048
#define YL_   2048
#define H_    768
#define OUT_  300
#define OUTP_ 304   // OUT padded to a multiple of 16 for B-fragment loads
#define K4H_  (4 * H_)

typedef __attribute__((ext_vector_type(16))) __bf16 v16bf_t;
typedef __attribute__((ext_vector_type(8)))  __bf16 v8bf_t;
typedef __attribute__((ext_vector_type(8)))  float  v8f_t;

// A-fragment (16x32 bf16): lane m (m = lane&15) holds row m.
// sel = lane>>4:  elements 0..7  -> K = sel*8 + t
//                 elements 8..15 -> K = 16 + sel*8 + t
__device__ __forceinline__ v16bf_t load_fragA(const __bf16* rowk, int sel) {
  v8bf_t lo = *(const v8bf_t*)(rowk + sel * 8);        // 16B load
  v8bf_t hi = *(const v8bf_t*)(rowk + 16 + sel * 8);   // 16B load
  v16bf_t f;
#pragma unroll
  for (int t = 0; t < 8; ++t) { f[t] = lo[t]; f[8 + t] = hi[t]; }
  return f;
}

__device__ __forceinline__ v8f_t wmma_bf16(v16bf_t a, v16bf_t b, v8f_t c) {
  return __builtin_amdgcn_wmma_f32_16x16x32_bf16(
      /*neg_a=*/false, a, /*neg_b=*/false, b,
      /*c_mod=*/(short)0, c, /*reuse_a=*/false, /*reuse_b=*/false);
}

// ---------------------------------------------------------------------------
// Prep: emb1 -> bf16, emb2*w_cq -> bf16
// ---------------------------------------------------------------------------
__global__ __launch_bounds__(256) void prep_bf16(const float* __restrict__ emb1,
                                                 const float* __restrict__ emb2,
                                                 const float* __restrict__ w_cq,
                                                 __bf16* __restrict__ emb1bf,
                                                 __bf16* __restrict__ emb2wbf) {
  size_t idx = (size_t)blockIdx.x * 256 + threadIdx.x;
  if (idx >= (size_t)B_ * XL_ * H_) return;
  int h = (int)(idx % H_);
  emb1bf[idx]  = (__bf16)emb1[idx];
  emb2wbf[idx] = (__bf16)(emb2[idx] * w_cq[h]);
}

// ---------------------------------------------------------------------------
// Tiled LDS transpose: emb1 (B,XL,H) f32 -> emb1T (B,H,XL) bf16
// ---------------------------------------------------------------------------
__global__ __launch_bounds__(256) void transpose_emb1(const float* __restrict__ emb1,
                                                      __bf16* __restrict__ emb1T) {
  __shared__ __bf16 tile[32][33];
  int b  = blockIdx.z;
  int i0 = blockIdx.x * 32;
  int h0 = blockIdx.y * 32;
  int tx = threadIdx.x;   // 0..31
  int ty = threadIdx.y;   // 0..7
  const float* src = emb1 + (size_t)b * XL_ * H_;
  __bf16* dst = emb1T + (size_t)b * H_ * XL_;
#pragma unroll
  for (int r = 0; r < 32; r += 8)
    tile[ty + r][tx] = (__bf16)src[(size_t)(i0 + ty + r) * H_ + h0 + tx];
  __syncthreads();
#pragma unroll
  for (int r = 0; r < 32; r += 8)
    dst[(size_t)(h0 + ty + r) * XL_ + i0 + tx] = tile[tx][ty + r];
}

// ---------------------------------------------------------------------------
// w_red (4H,OUT) f32 -> (4H,OUTP) bf16, zero-padded columns
// ---------------------------------------------------------------------------
__global__ __launch_bounds__(256) void conv_wred(const float* __restrict__ w_red,
                                                 __bf16* __restrict__ wredbf) {
  int idx = blockIdx.x * 256 + threadIdx.x;
  if (idx >= K4H_ * OUTP_) return;
  int k = idx / OUTP_;
  int o = idx % OUTP_;
  wredbf[idx] = (__bf16)((o < OUT_) ? w_red[(size_t)k * OUT_ + o] : 0.f);
}

// ---------------------------------------------------------------------------
// One wave per row: out[row] = dot(mat[row,:H], w) + bias
// ---------------------------------------------------------------------------
__global__ __launch_bounds__(256) void rowdot(const float* __restrict__ mat,
                                              const float* __restrict__ w,
                                              const float* __restrict__ bias,
                                              float* __restrict__ out, int rows) {
  int wid  = (int)((blockIdx.x * 256 + threadIdx.x) >> 5);
  int lane = threadIdx.x & 31;
  if (wid >= rows) return;
  const float* r = mat + (size_t)wid * H_;
  float s = 0.f;
  for (int h = lane; h < H_; h += 32) s += r[h] * w[h];
#pragma unroll
  for (int off = 16; off; off >>= 1) s += __shfl_xor(s, off, 32);
  if (lane == 0) out[wid] = s + bias[0];
}

// ---------------------------------------------------------------------------
// GEMM1: S[b,j,i] = sum_h emb2w[b,j,h]*emb1[b,i,h] + c[b,j] + q[b,i] + b_cq
// A = emb2w (row j, K=h contiguous); B = emb1T (lane k holds N=i contiguous).
// Block: 256 thr = 8 waves (2x4) -> 64(j) x 128(i) tile; wave = 32x32.
// ---------------------------------------------------------------------------
__global__ __launch_bounds__(256) void score_gemm(const __bf16* __restrict__ emb2wbf,
                                                  const __bf16* __restrict__ emb1T,
                                                  const float* __restrict__ c_score,
                                                  const float* __restrict__ q_score,
                                                  const float* __restrict__ b_cq,
                                                  float* __restrict__ S) {
  int b    = blockIdx.z;
  int wave = threadIdx.x >> 5;
  int lane = threadIdx.x & 31;
  int sel  = lane >> 4;
  int r16  = lane & 15;
  int jt = blockIdx.y * 64  + (wave >> 2) * 32;   // wave M base (32 rows)
  int it = blockIdx.x * 128 + (wave & 3) * 32;    // wave N base (32 cols)

  const __bf16* arow0 = emb2wbf + (size_t)(b * YL_ + jt + r16) * H_;
  const __bf16* arow1 = arow0 + (size_t)16 * H_;
  const __bf16* Tb    = emb1T + (size_t)b * H_ * XL_;

  v8f_t acc00 = {}, acc01 = {}, acc10 = {}, acc11 = {};
#pragma unroll 2
  for (int kb = 0; kb < H_; kb += 32) {
    v16bf_t a0 = load_fragA(arow0 + kb, sel);
    v16bf_t a1 = load_fragA(arow1 + kb, sel);
    const __bf16* bp = Tb + (size_t)kb * XL_ + it + r16;
    v16bf_t b0, b1;
#pragma unroll
    for (int t = 0; t < 8; ++t) {
      size_t k0 = (size_t)(sel * 8 + t) * XL_, k1 = (size_t)(16 + sel * 8 + t) * XL_;
      b0[t] = bp[k0]; b0[8 + t] = bp[k1];
      b1[t] = bp[k0 + 16]; b1[8 + t] = bp[k1 + 16];
    }
    acc00 = wmma_bf16(a0, b0, acc00);
    acc01 = wmma_bf16(a0, b1, acc01);
    acc10 = wmma_bf16(a1, b0, acc10);
    acc11 = wmma_bf16(a1, b1, acc11);
  }
  // C layout: lane&15 = N, sel*8 + r = M
  int col0 = it + r16;
  int col1 = col0 + 16;
  float bq = b_cq[0];
  float q0 = q_score[b * XL_ + col0] + bq;
  float q1 = q_score[b * XL_ + col1] + bq;
  float* Srow = S + (size_t)b * YL_ * XL_;
#pragma unroll
  for (int r = 0; r < 8; ++r) {
    int j0 = jt + sel * 8 + r;
    int j1 = j0 + 16;
    float c0 = c_score[b * YL_ + j0];
    float c1 = c_score[b * YL_ + j1];
    Srow[(size_t)j0 * XL_ + col0] = acc00[r] + c0 + q0;
    Srow[(size_t)j0 * XL_ + col1] = acc01[r] + c0 + q1;
    Srow[(size_t)j1 * XL_ + col0] = acc10[r] + c1 + q0;
    Srow[(size_t)j1 * XL_ + col1] = acc11[r] + c1 + q1;
  }
}

// ---------------------------------------------------------------------------
// Row softmax over i (XL): writes bf16 probs and the row max (for b_att)
// ---------------------------------------------------------------------------
__global__ __launch_bounds__(256) void softmax_rows(const float* __restrict__ S,
                                                    __bf16* __restrict__ A,
                                                    float* __restrict__ rowmax) {
  int row = blockIdx.x;                 // over B*YL
  const float* s = S + (size_t)row * XL_;
  __bf16* a = A + (size_t)row * XL_;
  __shared__ float red[8];
  int tid = threadIdx.x, lane = tid & 31, wid = tid >> 5;

  float m = -3.402823e38f;
  for (int i = tid; i < XL_; i += 256) m = fmaxf(m, s[i]);
#pragma unroll
  for (int off = 16; off; off >>= 1) m = fmaxf(m, __shfl_xor(m, off, 32));
  if (lane == 0) red[wid] = m;
  __syncthreads();
  m = fmaxf(fmaxf(fmaxf(red[0], red[1]), fmaxf(red[2], red[3])),
            fmaxf(fmaxf(red[4], red[5]), fmaxf(red[6], red[7])));
  __syncthreads();

  float sum = 0.f;
  for (int i = tid; i < XL_; i += 256) {
    float e = __expf(s[i] - m);
    a[i] = (__bf16)e;
    sum += e;
  }
#pragma unroll
  for (int off = 16; off; off >>= 1) sum += __shfl_xor(sum, off, 32);
  if (lane == 0) red[wid] = sum;
  __syncthreads();
  sum = red[0] + red[1] + red[2] + red[3] + red[4] + red[5] + red[6] + red[7];
  float inv = 1.f / sum;
  for (int i = tid; i < XL_; i += 256) a[i] = (__bf16)((float)a[i] * inv);
  if (tid == 0) rowmax[row] = m;
}

// ---------------------------------------------------------------------------
// b_att = softmax_j(rowmax[b,:]) ; one block per batch
// ---------------------------------------------------------------------------
__global__ __launch_bounds__(256) void batt_softmax(const float* __restrict__ rowmax,
                                                    float* __restrict__ b_att) {
  int b = blockIdx.x;
  const float* s = rowmax + (size_t)b * YL_;
  float* o = b_att + (size_t)b * YL_;
  __shared__ float red[8];
  int tid = threadIdx.x, lane = tid & 31, wid = tid >> 5;

  float m = -3.402823e38f;
  for (int i = tid; i < YL_; i += 256) m = fmaxf(m, s[i]);
#pragma unroll
  for (int off = 16; off; off >>= 1) m = fmaxf(m, __shfl_xor(m, off, 32));
  if (lane == 0) red[wid] = m;
  __syncthreads();
  m = fmaxf(fmaxf(fmaxf(red[0], red[1]), fmaxf(red[2], red[3])),
            fmaxf(fmaxf(red[4], red[5]), fmaxf(red[6], red[7])));
  __syncthreads();

  float sum = 0.f;
  for (int i = tid; i < YL_; i += 256) { float e = __expf(s[i] - m); o[i] = e; sum += e; }
#pragma unroll
  for (int off = 16; off; off >>= 1) sum += __shfl_xor(sum, off, 32);
  if (lane == 0) red[wid] = sum;
  __syncthreads();
  sum = red[0] + red[1] + red[2] + red[3] + red[4] + red[5] + red[6] + red[7];
  float inv = 1.f / sum;
  for (int i = tid; i < YL_; i += 256) o[i] *= inv;
}

// ---------------------------------------------------------------------------
// x2y[b,h] = sum_j b_att[b,j] * emb2[b,j,h]
// ---------------------------------------------------------------------------
__global__ __launch_bounds__(256) void x2y_kernel(const float* __restrict__ emb2,
                                                  const float* __restrict__ b_att,
                                                  float* __restrict__ x2y) {
  int b = blockIdx.y;
  int h = blockIdx.x * 256 + threadIdx.x;
  if (h >= H_) return;
  const float* e = emb2 + (size_t)b * YL_ * H_ + h;
  const float* w = b_att + (size_t)b * YL_;
  float acc = 0.f;
  for (int j = 0; j < YL_; ++j) acc += w[j] * e[(size_t)j * H_];
  x2y[b * H_ + h] = acc;
}

// ---------------------------------------------------------------------------
// GEMM2: y2x[b,j,h] = sum_i a[b,j,i] * emb1[b,i,h]
// A = probs (row j, K=i contiguous); B = emb1bf (lane k=i holds N=h contiguous)
// Block: 8 waves (2x4) -> 64(j) x 128(h) tile; wave = 32x32.
// ---------------------------------------------------------------------------
__global__ __launch_bounds__(256) void att_gemm(const __bf16* __restrict__ probs,
                                                const __bf16* __restrict__ emb1bf,
                                                float* __restrict__ y2x) {
  int b    = blockIdx.z;
  int wave = threadIdx.x >> 5;
  int lane = threadIdx.x & 31;
  int sel  = lane >> 4;
  int r16  = lane & 15;
  int jt = blockIdx.y * 64  + (wave >> 2) * 32;
  int ht = blockIdx.x * 128 + (wave & 3) * 32;

  const __bf16* arow0 = probs + (size_t)(b * YL_ + jt + r16) * XL_;
  const __bf16* arow1 = arow0 + (size_t)16 * XL_;
  const __bf16* Bb    = emb1bf + (size_t)b * XL_ * H_;

  v8f_t acc00 = {}, acc01 = {}, acc10 = {}, acc11 = {};
#pragma unroll 2
  for (int kb = 0; kb < XL_; kb += 32) {
    v16bf_t a0 = load_fragA(arow0 + kb, sel);
    v16bf_t a1 = load_fragA(arow1 + kb, sel);
    const __bf16* bp = Bb + (size_t)kb * H_ + ht + r16;
    v16bf_t b0, b1;
#pragma unroll
    for (int t = 0; t < 8; ++t) {
      size_t k0 = (size_t)(sel * 8 + t) * H_, k1 = (size_t)(16 + sel * 8 + t) * H_;
      b0[t] = bp[k0]; b0[8 + t] = bp[k1];
      b1[t] = bp[k0 + 16]; b1[8 + t] = bp[k1 + 16];
    }
    acc00 = wmma_bf16(a0, b0, acc00);
    acc01 = wmma_bf16(a0, b1, acc01);
    acc10 = wmma_bf16(a1, b0, acc10);
    acc11 = wmma_bf16(a1, b1, acc11);
  }
  int col0 = ht + r16;
  int col1 = col0 + 16;
  float* yrow = y2x + (size_t)b * YL_ * H_;
#pragma unroll
  for (int r = 0; r < 8; ++r) {
    int j0 = jt + sel * 8 + r;
    int j1 = j0 + 16;
    yrow[(size_t)j0 * H_ + col0] = acc00[r];
    yrow[(size_t)j0 * H_ + col1] = acc01[r];
    yrow[(size_t)j1 * H_ + col0] = acc10[r];
    yrow[(size_t)j1 * H_ + col1] = acc11[r];
  }
}

// ---------------------------------------------------------------------------
// GEMM3: out[m,o] = sum_k z[m,k]*w_red[k,o] + b_red[o], m over B*YL
// z row built on the fly: [emb2 | y2x | emb2*y2x | emb2*x2y]
// ---------------------------------------------------------------------------
__device__ __forceinline__ v16bf_t build_z_frag(const float* __restrict__ emb2row,
                                                const float* __restrict__ y2xrow,
                                                const float* __restrict__ x2yb,
                                                int kbase, int sel) {
  int region = kbase / H_;
  int hk = kbase - region * H_;
  int o0 = hk + sel * 8;
  int o1 = hk + 16 + sel * 8;
  v16bf_t f;
#pragma unroll
  for (int t = 0; t < 8; ++t) {
    int h0 = o0 + t, h1 = o1 + t;
    float v0, v1;
    if (region == 0)      { v0 = emb2row[h0];              v1 = emb2row[h1]; }
    else if (region == 1) { v0 = y2xrow[h0];               v1 = y2xrow[h1]; }
    else if (region == 2) { v0 = emb2row[h0] * y2xrow[h0]; v1 = emb2row[h1] * y2xrow[h1]; }
    else                  { v0 = emb2row[h0] * x2yb[h0];   v1 = emb2row[h1] * x2yb[h1]; }
    f[t] = (__bf16)v0; f[8 + t] = (__bf16)v1;
  }
  return f;
}

__global__ __launch_bounds__(256) void reduce_gemm(const float* __restrict__ emb2,
                                                   const float* __restrict__ y2x,
                                                   const float* __restrict__ x2y,
                                                   const __bf16* __restrict__ wredbf,
                                                   const float* __restrict__ b_red,
                                                   float* __restrict__ out) {
  int wave = threadIdx.x >> 5;
  int lane = threadIdx.x & 31;
  int sel  = lane >> 4;
  int r16  = lane & 15;
  int mt = (blockIdx.y * 8 + wave) * 16;   // row tile in flattened (B*YL)
  int ot = blockIdx.x * 16;                // output tile (padded N)

  int mrow = mt + r16;                     // this lane's A row
  int b = mrow >> 11;                      // YL == 2048
  const float* emb2row = emb2 + (size_t)mrow * H_;
  const float* y2xrow  = y2x  + (size_t)mrow * H_;
  const float* x2yb    = x2y  + (size_t)b * H_;

  v8f_t acc = {};
  for (int kb = 0; kb < K4H_; kb += 32) {
    v16bf_t af = build_z_frag(emb2row, y2xrow, x2yb, kb, sel);
    const __bf16* bp = wredbf + (size_t)kb * OUTP_ + ot + r16;
    v16bf_t bf;
#pragma unroll
    for (int t = 0; t < 8; ++t) {
      bf[t] = bp[(size_t)(sel * 8 + t) * OUTP_];
      bf[8 + t] = bp[(size_t)(16 + sel * 8 + t) * OUTP_];
    }
    acc = wmma_bf16(af, bf, acc);
  }
  int col  = ot + r16;
  int rowb = mt + sel * 8;
  if (col < OUT_) {
    float bo = b_red[col];
#pragma unroll
    for (int r = 0; r < 8; ++r)
      out[(size_t)(rowb + r) * OUT_ + col] = acc[r] + bo;
  }
}

// ---------------------------------------------------------------------------
// Launch
// ---------------------------------------------------------------------------
extern "C" void kernel_launch(void* const* d_in, const int* in_sizes, int n_in,
                              void* d_out, int out_size, void* d_ws, size_t ws_size,
                              hipStream_t stream) {
  (void)in_sizes; (void)n_in; (void)out_size; (void)ws_size;
  const float* emb1  = (const float*)d_in[0];
  const float* emb2  = (const float*)d_in[1];
  const float* w_c   = (const float*)d_in[2];
  const float* b_c   = (const float*)d_in[3];
  const float* w_q   = (const float*)d_in[4];
  const float* b_q   = (const float*)d_in[5];
  const float* w_cq  = (const float*)d_in[6];
  const float* b_cq  = (const float*)d_in[7];
  const float* w_red = (const float*)d_in[8];
  const float* b_red = (const float*)d_in[9];
  float* out = (float*)d_out;

  // ---- carve workspace ----
  char* ws = (char*)d_ws;
  size_t off = 0;
  auto carve = [&](size_t bytes) {
    void* p = ws + off;
    off = (off + bytes + 255) & ~(size_t)255;
    return p;
  };
  const size_t NE = (size_t)B_ * XL_ * H_;          // == B*YL*H
  __bf16* emb1bf  = (__bf16*)carve(NE * 2);
  __bf16* emb2wbf = (__bf16*)carve(NE * 2);
  __bf16* emb1T   = (__bf16*)carve(NE * 2);
  __bf16* wredbf  = (__bf16*)carve((size_t)K4H_ * OUTP_ * 2);
  float*  q_score = (float*)carve((size_t)B_ * XL_ * 4);
  float*  c_score = (float*)carve((size_t)B_ * YL_ * 4);
  float*  S       = (float*)carve((size_t)B_ * YL_ * XL_ * 4);
  __bf16* probs   = (__bf16*)carve((size_t)B_ * YL_ * XL_ * 2);
  float*  rowmax  = (float*)carve((size_t)B_ * YL_ * 4);
  float*  b_att   = (float*)carve((size_t)B_ * YL_ * 4);
  float*  x2y     = (float*)carve((size_t)B_ * H_ * 4);
  float*  y2x     = (float*)carve((size_t)B_ * YL_ * H_ * 4);

  // 1) bf16 conversions
  {
    int nblk = (int)((NE + 255) / 256);
    prep_bf16<<<nblk, 256, 0, stream>>>(emb1, emb2, w_cq, emb1bf, emb2wbf);
  }
  // 2) emb1 transpose -> (B,H,XL) bf16
  transpose_emb1<<<dim3(XL_ / 32, H_ / 32, B_), dim3(32, 8), 0, stream>>>(emb1, emb1T);
  // 3) w_red -> bf16 padded
  conv_wred<<<(K4H_ * OUTP_ + 255) / 256, 256, 0, stream>>>(w_red, wredbf);
  // 4) c_score / q_score (wave per row)
  rowdot<<<(B_ * YL_) / 8, 256, 0, stream>>>(emb2, w_c, b_c, c_score, B_ * YL_);
  rowdot<<<(B_ * XL_) / 8, 256, 0, stream>>>(emb1, w_q, b_q, q_score, B_ * XL_);
  // 5) score GEMM (WMMA, 32x32 per wave) + bias fusion
  score_gemm<<<dim3(XL_ / 128, YL_ / 64, B_), 256, 0, stream>>>(
      emb2wbf, emb1T, c_score, q_score, b_cq, S);
  // 6) row softmax -> bf16 probs + rowmax
  softmax_rows<<<B_ * YL_, 256, 0, stream>>>(S, probs, rowmax);
  // 7) b_att softmax over rows
  batt_softmax<<<B_, 256, 0, stream>>>(rowmax, b_att);
  // 8) x2y GEMV
  x2y_kernel<<<dim3(H_ / 256, B_), 256, 0, stream>>>(emb2, b_att, x2y);
  // 9) attention GEMM (WMMA, 32x32 per wave)
  att_gemm<<<dim3(H_ / 128, YL_ / 64, B_), 256, 0, stream>>>(probs, emb1bf, y2x);
  // 10) reduction GEMM (WMMA) with fused z-construction + bias
  reduce_gemm<<<dim3(OUTP_ / 16, (B_ * YL_) / (16 * 8)), 256, 0, stream>>>(
      emb2, y2x, x2y, wredbf, b_red, out);
}
